// EncoderLayer_28509992910939
// MI455X (gfx1250) — matrix operand order repaired and checked by reading, not compiled, not run on hardware
//
#include <hip/hip_runtime.h>
#include <hip/hip_bf16.h>

typedef __attribute__((ext_vector_type(16))) _Float16 v16h;
typedef __attribute__((ext_vector_type(8)))  _Float16 v8h;
typedef __attribute__((ext_vector_type(4)))  _Float16 v4h;
typedef __attribute__((ext_vector_type(8)))  float    v8f;
typedef __attribute__((ext_vector_type(4)))  int      v4i;

#define DM    768
#define DFF   2048
#define NH    12
#define DK    64
#define SEQ   2048
#define BATCH 4
#define MROWS (BATCH * SEQ)   // 8192

#if defined(__AMDGCN__) && __has_builtin(__builtin_amdgcn_global_load_async_to_lds_b128) && \
    __has_builtin(__builtin_amdgcn_s_wait_asynccnt)
#define USE_ASYNC_LDS 1
#else
#define USE_ASYNC_LDS 0
#endif

__device__ __forceinline__ v8f zero8f() {
  v8f z;
#pragma unroll
  for (int i = 0; i < 8; ++i) z[i] = 0.0f;
  return z;
}

__device__ __forceinline__ v16h cat8(v8h lo, v8h hi) {
  return __builtin_shufflevector(lo, hi, 0, 1, 2, 3, 4, 5, 6, 7,
                                 8, 9, 10, 11, 12, 13, 14, 15);
}

__device__ __forceinline__ v8f wmma16(v16h a, v16h b, v8f c) {
  // D = A(16x32 f16) * B(32x16 f16) + C(16x16 f32)
  return __builtin_amdgcn_wmma_f32_16x16x32_f16(false, a, false, b,
                                                (short)0, c, false, false);
}

#if USE_ASYNC_LDS
// Per-lane 16-byte async DMA: LDS[l] = MEM[g], tracked with ASYNCcnt.
// Builtin takes typed v4i pointers (global AS1 source, LDS AS3 dest).
__device__ __forceinline__ void async_copy16(const _Float16* g, _Float16* l) {
  __builtin_amdgcn_global_load_async_to_lds_b128(
      (__attribute__((address_space(1))) v4i*)(unsigned long long)(const void*)g,
      (__attribute__((address_space(3))) v4i*)(unsigned int)(unsigned long long)(void*)l,
      0, 0);
}
#endif

// ---------------------------------------------------------------------------
// fp32 -> fp16 conversion
// ---------------------------------------------------------------------------
__global__ __launch_bounds__(256) void f32_to_f16_kernel(
    const float* __restrict__ in, _Float16* __restrict__ out, int n) {
  int i = (blockIdx.x * 256 + threadIdx.x) * 4;
  if (i + 3 < n) {
    float4 f = *(const float4*)(in + i);
    v4h h;
    h[0] = (_Float16)f.x; h[1] = (_Float16)f.y;
    h[2] = (_Float16)f.z; h[3] = (_Float16)f.w;
    *(v4h*)(out + i) = h;
  }
}

// ---------------------------------------------------------------------------
// WMMA GEMM: C[M,N] = A[M,K](f16) @ B[K,N](f16) + bias, optional ReLU,
// f32 and/or f16 stores. Block tile 128M x 128N, 8 waves (4Mx2N),
// wave tile 32M x 64N (8 WMMAs / K-step), K-step 32, async LDS staging.
// ---------------------------------------------------------------------------
template <bool F32OUT, bool F16OUT, bool RELU>
__global__ __launch_bounds__(256) void gemm_f16_kernel(
    const _Float16* __restrict__ A, const _Float16* __restrict__ Bm,
    const float* __restrict__ bias, float* __restrict__ Cf,
    _Float16* __restrict__ Ch, int N, int K) {
  __shared__ __align__(32) _Float16 As[128 * 32];  // [row][k]
  __shared__ __align__(32) _Float16 Bs[32 * 128];  // [k][col]

  const int tid   = threadIdx.x;
  const int lane  = tid & 31;
  const int wave  = tid >> 5;
  const int waveM = wave & 3;
  const int waveN = wave >> 2;
  const int colb  = lane & 15;
  const int hi    = lane >> 4;

  const int bm = blockIdx.x * 128;
  const int bn = blockIdx.y * 128;

  v8f acc[2][4];
#pragma unroll
  for (int mi = 0; mi < 2; ++mi)
#pragma unroll
    for (int t = 0; t < 4; ++t) acc[mi][t] = zero8f();

  const int arow0 = waveM * 32 + colb;  // first A-fragment row for this lane
  const int koff  = hi * 8;             // A-fragment K sub-offset per half
  const int brow  = colb + hi * 16;     // B-fragment K row for this lane

  for (int k0 = 0; k0 < K; k0 += 32) {
    __syncthreads();
#if USE_ASYNC_LDS
    // A tile 128x32 halves = 512 x 16B chunks; B tile 32x128 = 512 chunks.
#pragma unroll
    for (int i = 0; i < 2; ++i) {
      int id = tid + i * 256;               // 0..511
      int ra = id >> 2, ca = (id & 3) * 8;  // A: 4 chunks per row
      async_copy16(&A[(size_t)(bm + ra) * K + k0 + ca], &As[ra * 32 + ca]);
      int rb = id >> 4, cb = (id & 15) * 8; // B: 16 chunks per row
      async_copy16(&Bm[(size_t)(k0 + rb) * N + bn + cb], &Bs[rb * 128 + cb]);
    }
    __builtin_amdgcn_s_wait_asynccnt(0);
#else
#pragma unroll
    for (int i = 0; i < 8; ++i) {
      int idx = tid + i * 256;              // A: 128x16 dwords
      int r = idx >> 4;
      int c = (idx & 15) * 2;
      *(unsigned*)&As[r * 32 + c] =
          *(const unsigned*)&A[(size_t)(bm + r) * K + k0 + c];
    }
#pragma unroll
    for (int i = 0; i < 8; ++i) {
      int idx = tid + i * 256;              // B: 32x64 dwords
      int r = idx >> 6;
      int c = (idx & 63) * 2;
      *(unsigned*)&Bs[r * 128 + c] =
          *(const unsigned*)&Bm[(size_t)(k0 + r) * N + bn + c];
    }
#endif
    __syncthreads();

    if (k0 + 32 < K) {  // prefetch next tiles into cache
      __builtin_prefetch(&A[(size_t)(bm + (tid >> 1)) * K + k0 + 32], 0, 1);
      __builtin_prefetch(&Bm[(size_t)(k0 + 32 + (tid >> 3)) * N + bn + (tid & 7) * 16], 0, 1);
    }

    // A fragments: ISA 16-bit A 16x32 layout (two contiguous 8-half runs).
    v16h af0 = cat8(*(const v8h*)&As[arow0 * 32 + koff],
                    *(const v8h*)&As[arow0 * 32 + 16 + koff]);
    v16h af1 = cat8(*(const v8h*)&As[(arow0 + 16) * 32 + koff],
                    *(const v8h*)&As[(arow0 + 16) * 32 + 16 + koff]);
#pragma unroll
    for (int t = 0; t < 4; ++t) {
      // B fragment: lane = N column, 16 K halves in table order.
      v16h bf;
#pragma unroll
      for (int s = 0; s < 16; ++s)
        bf[s] = Bs[(koff + (s & 7) + (s >> 3) * 16) * 128 + waveN * 64 + t * 16 + colb];
      acc[0][t] = wmma16(af0, bf, acc[0][t]);
      acc[1][t] = wmma16(af1, bf, acc[1][t]);
    }
  }

#pragma unroll
  for (int mi = 0; mi < 2; ++mi)
#pragma unroll
    for (int t = 0; t < 4; ++t) {
      const int col = bn + waveN * 64 + t * 16 + colb;
      const float bv = bias[col];
#pragma unroll
      for (int r = 0; r < 8; ++r) {
        const int row = bm + waveM * 32 + mi * 16 + r + 8 * hi;
        float v = acc[mi][t][r] + bv;
        if (RELU) v = fmaxf(v, 0.0f);
        if (F32OUT) Cf[(size_t)row * N + col] = v;
        if (F16OUT) Ch[(size_t)row * N + col] = (_Float16)v;
      }
    }
}

// ---------------------------------------------------------------------------
// Flash attention: per (b,h), 8 waves x 16 query rows per block, stream over
// keys in tiles of 32 with online softmax. Q/K/V layout: [B,S,H*DK] f16.
// ---------------------------------------------------------------------------
__global__ __launch_bounds__(256) void attn_kernel(
    const _Float16* __restrict__ Qh, const _Float16* __restrict__ Kh,
    const _Float16* __restrict__ Vh, const float* __restrict__ mask,
    _Float16* __restrict__ Oh) {
  __shared__ __align__(32) _Float16 Kt[DK * 32];     // transposed: [kd][key]
  __shared__ __align__(32) _Float16 Vs[32 * DK];     // [key][d]
  __shared__ __align__(32) _Float16 Ps[8 * 16 * 32]; // per-wave P tiles

  const int tid  = threadIdx.x;
  const int lane = tid & 31;
  const int wave = tid >> 5;
  const int colb = lane & 15;
  const int hi   = lane >> 4;

  const int bh = blockIdx.y;
  const int b  = bh / NH;
  const int h  = bh - b * NH;
  const size_t base = (size_t)b * SEQ * DM + (size_t)h * DK;

  const int q0 = blockIdx.x * 128 + wave * 16;
  const int koff = hi * 8;
  const int brow = colb + hi * 16;  // 0..31

  // Q fragments (A-operand), loaded once: kdim 0..31 and 32..63.
  const _Float16* qp = Qh + base + (size_t)(q0 + colb) * DM;
  v16h qf0 = cat8(*(const v8h*)(qp + koff), *(const v8h*)(qp + 16 + koff));
  v16h qf1 = cat8(*(const v8h*)(qp + 32 + koff), *(const v8h*)(qp + 48 + koff));

  float m[8], l[8];
  v8f o[4];
#pragma unroll
  for (int r = 0; r < 8; ++r) { m[r] = -3.0e38f; l[r] = 0.0f; }
#pragma unroll
  for (int t = 0; t < 4; ++t) o[t] = zero8f();

  _Float16* Psw = &Ps[wave * 512];
  const _Float16* Ppr = &Ps[wave * 512 + colb * 32];

  for (int kt = 0; kt < SEQ; kt += 32) {
    __syncthreads();
    // Stage K transposed: Kt[kd][key]. 1024 dwords, 4 per thread.
#pragma unroll
    for (int i = 0; i < 4; ++i) {
      int idx = tid + i * 256;
      int key = idx >> 5;
      int dd  = (idx & 31) * 2;
      union { unsigned u; _Float16 f[2]; } w;
      w.u = *(const unsigned*)&Kh[base + (size_t)(kt + key) * DM + dd];
      Kt[dd * 32 + key]       = w.f[0];
      Kt[(dd + 1) * 32 + key] = w.f[1];
    }
    // Stage V row-major: Vs[key][d].
#pragma unroll
    for (int i = 0; i < 4; ++i) {
      int idx = tid + i * 256;
      int key = idx >> 5;
      int dd  = (idx & 31) * 2;
      *(unsigned*)&Vs[key * DK + dd] =
          *(const unsigned*)&Vh[base + (size_t)(kt + key) * DM + dd];
    }
    __syncthreads();

    // Scores S(16x32) = Q(16x64) @ K^T : 4 WMMAs.
    v8f s0 = zero8f(), s1 = zero8f();
    {
      v16h b00, b01, b10, b11;
#pragma unroll
      for (int s = 0; s < 16; ++s) {
        const int kk = koff + (s & 7) + (s >> 3) * 16;
        b00[s] = Kt[kk * 32 + colb];
        b01[s] = Kt[kk * 32 + 16 + colb];
        b10[s] = Kt[(32 + kk) * 32 + colb];
        b11[s] = Kt[(32 + kk) * 32 + 16 + colb];
      }
      s0 = wmma16(qf0, b00, s0);
      s0 = wmma16(qf1, b10, s0);
      s1 = wmma16(qf0, b01, s1);
      s1 = wmma16(qf1, b11, s1);
    }

    const float mk0 = mask[(size_t)b * SEQ + kt + colb] * -1.0e9f;
    const float mk1 = mask[(size_t)b * SEQ + kt + 16 + colb] * -1.0e9f;

    float p0a[8], p1a[8];
#pragma unroll
    for (int r = 0; r < 8; ++r) {
      float a0 = s0[r] * 0.125f + mk0;   // 1/sqrt(64)
      float a1 = s1[r] * 0.125f + mk1;
      float tm = fmaxf(a0, a1);
#pragma unroll
      for (int off = 1; off < 16; off <<= 1)
        tm = fmaxf(tm, __shfl_xor(tm, off, 32));
      float mn   = fmaxf(m[r], tm);
      float corr = __expf(m[r] - mn);
      float p0   = __expf(a0 - mn);
      float p1   = __expf(a1 - mn);
      float rs   = p0 + p1;
#pragma unroll
      for (int off = 1; off < 16; off <<= 1)
        rs += __shfl_xor(rs, off, 32);
      l[r] = l[r] * corr + rs;
      m[r] = mn;
      p0a[r] = p0;
      p1a[r] = p1;
#pragma unroll
      for (int t = 0; t < 4; ++t) o[t][r] *= corr;
    }

    // C-layout -> A-layout via per-wave LDS round trip (DS is in-order per wave).
#pragma unroll
    for (int r = 0; r < 8; ++r) {
      Psw[(r + 8 * hi) * 32 + colb]      = (_Float16)p0a[r];
      Psw[(r + 8 * hi) * 32 + 16 + colb] = (_Float16)p1a[r];
    }
    v16h pf = cat8(*(const v8h*)(Ppr + koff), *(const v8h*)(Ppr + 16 + koff));

    // O(16x64) += P(16x32) @ V(32x64) : 4 WMMAs.
#pragma unroll
    for (int t = 0; t < 4; ++t) {
      v16h vf;
#pragma unroll
      for (int s = 0; s < 16; ++s)
        vf[s] = Vs[(koff + (s & 7) + (s >> 3) * 16) * DK + t * 16 + colb];
      o[t] = wmma16(pf, vf, o[t]);
    }
  }

  float inv[8];
#pragma unroll
  for (int r = 0; r < 8; ++r) inv[r] = 1.0f / l[r];
#pragma unroll
  for (int t = 0; t < 4; ++t)
#pragma unroll
    for (int r = 0; r < 8; ++r) {
      size_t idx = base + (size_t)(q0 + r + 8 * hi) * DM + t * 16 + colb;
      Oh[idx] = (_Float16)(o[t][r] * inv[r]);
    }
}

// ---------------------------------------------------------------------------
// y = LayerNorm(a + b) * gamma + beta  (population variance, eps = 1e-3).
// Writes f32 always, f16 copy if yh != nullptr. One block per row, D = 768.
// ---------------------------------------------------------------------------
__global__ __launch_bounds__(256) void add_ln_kernel(
    const float* __restrict__ a, const float* __restrict__ bsrc,
    const float* __restrict__ gamma, const float* __restrict__ beta,
    float* __restrict__ yf, _Float16* __restrict__ yh) {
  const int row = blockIdx.x;
  const float* ap = a + (size_t)row * DM;
  const float* bp = bsrc + (size_t)row * DM;

  float v[3];
  float s = 0.0f, ss = 0.0f;
#pragma unroll
  for (int i = 0; i < 3; ++i) {
    int c = threadIdx.x + i * 256;
    float t = ap[c] + bp[c];
    v[i] = t;
    s += t;
    ss += t * t;
  }
#pragma unroll
  for (int off = 16; off >= 1; off >>= 1) {
    s  += __shfl_xor(s, off, 32);
    ss += __shfl_xor(ss, off, 32);
  }
  __shared__ float sw[8], ssw[8];
  const int lane = threadIdx.x & 31, wave = threadIdx.x >> 5;
  if (lane == 0) { sw[wave] = s; ssw[wave] = ss; }
  __syncthreads();
  if (wave == 0) {
    float s2  = (lane < 8) ? sw[lane]  : 0.0f;
    float ss2 = (lane < 8) ? ssw[lane] : 0.0f;
#pragma unroll
    for (int off = 4; off >= 1; off >>= 1) {
      s2  += __shfl_xor(s2, off, 32);
      ss2 += __shfl_xor(ss2, off, 32);
    }
    if (lane == 0) { sw[0] = s2; ssw[0] = ss2; }
  }
  __syncthreads();
  const float mu  = sw[0] * (1.0f / DM);
  const float var = ssw[0] * (1.0f / DM) - mu * mu;
  const float rs  = rsqrtf(var + 1e-3f);
#pragma unroll
  for (int i = 0; i < 3; ++i) {
    int c = threadIdx.x + i * 256;
    float y = (v[i] - mu) * rs * gamma[c] + beta[c];
    yf[(size_t)row * DM + c] = y;
    if (yh) yh[(size_t)row * DM + c] = (_Float16)y;
  }
}

// ---------------------------------------------------------------------------
extern "C" void kernel_launch(void* const* d_in, const int* in_sizes, int n_in,
                              void* d_out, int out_size, void* d_ws,
                              size_t ws_size, hipStream_t stream) {
  (void)in_sizes; (void)n_in; (void)out_size; (void)ws_size;
  const float* x    = (const float*)d_in[0];
  const float* mask = (const float*)d_in[1];
  const float* Wq = (const float*)d_in[2];  const float* bq = (const float*)d_in[3];
  const float* Wk = (const float*)d_in[4];  const float* bk = (const float*)d_in[5];
  const float* Wv = (const float*)d_in[6];  const float* bv = (const float*)d_in[7];
  const float* Wo = (const float*)d_in[8];  const float* bo = (const float*)d_in[9];
  const float* W1 = (const float*)d_in[10]; const float* b1 = (const float*)d_in[11];
  const float* W2 = (const float*)d_in[12]; const float* b2 = (const float*)d_in[13];
  const float* g1 = (const float*)d_in[14]; const float* be1 = (const float*)d_in[15];
  const float* g2 = (const float*)d_in[16]; const float* be2 = (const float*)d_in[17];
  float* out = (float*)d_out;

  char* w = (char*)d_ws;
  size_t off = 0;
  auto take = [&](size_t bytes) -> char* {
    char* p = w + off;
    off += (bytes + 255) & ~(size_t)255;
    return p;
  };

  _Float16* xh    = (_Float16*)take((size_t)MROWS * DM * 2);
  _Float16* Wqh   = (_Float16*)take((size_t)DM * DM * 2);
  _Float16* Wkh   = (_Float16*)take((size_t)DM * DM * 2);
  _Float16* Wvh   = (_Float16*)take((size_t)DM * DM * 2);
  _Float16* Woh   = (_Float16*)take((size_t)DM * DM * 2);
  _Float16* W1h   = (_Float16*)take((size_t)DM * DFF * 2);
  _Float16* W2h   = (_Float16*)take((size_t)DFF * DM * 2);
  _Float16* Qh    = (_Float16*)take((size_t)MROWS * DM * 2);
  _Float16* Kh    = (_Float16*)take((size_t)MROWS * DM * 2);
  _Float16* Vh    = (_Float16*)take((size_t)MROWS * DM * 2);
  _Float16* attnh = (_Float16*)take((size_t)MROWS * DM * 2);
  float*    attno = (float*)take((size_t)MROWS * DM * 4);
  float*    x1    = (float*)take((size_t)MROWS * DM * 4);
  _Float16* x1h   = (_Float16*)take((size_t)MROWS * DM * 2);
  _Float16* hh    = (_Float16*)take((size_t)MROWS * DFF * 2);
  float*    ff    = (float*)take((size_t)MROWS * DM * 4);

  // fp32 -> fp16 conversions
  f32_to_f16_kernel<<<(MROWS * DM) / 1024, 256, 0, stream>>>(x, xh, MROWS * DM);
  f32_to_f16_kernel<<<(DM * DM) / 1024, 256, 0, stream>>>(Wq, Wqh, DM * DM);
  f32_to_f16_kernel<<<(DM * DM) / 1024, 256, 0, stream>>>(Wk, Wkh, DM * DM);
  f32_to_f16_kernel<<<(DM * DM) / 1024, 256, 0, stream>>>(Wv, Wvh, DM * DM);
  f32_to_f16_kernel<<<(DM * DM) / 1024, 256, 0, stream>>>(Wo, Woh, DM * DM);
  f32_to_f16_kernel<<<(DM * DFF) / 1024, 256, 0, stream>>>(W1, W1h, DM * DFF);
  f32_to_f16_kernel<<<(DFF * DM) / 1024, 256, 0, stream>>>(W2, W2h, DFF * DM);

  dim3 g768(MROWS / 128, DM / 128);   // 64 x 6
  dim3 gff(MROWS / 128, DFF / 128);   // 64 x 16

  // Q/K/V projections (f16 out)
  gemm_f16_kernel<false, true, false><<<g768, 256, 0, stream>>>(
      xh, Wqh, bq, nullptr, Qh, DM, DM);
  gemm_f16_kernel<false, true, false><<<g768, 256, 0, stream>>>(
      xh, Wkh, bk, nullptr, Kh, DM, DM);
  gemm_f16_kernel<false, true, false><<<g768, 256, 0, stream>>>(
      xh, Wvh, bv, nullptr, Vh, DM, DM);

  // Flash attention
  attn_kernel<<<dim3(SEQ / 128, BATCH * NH), 256, 0, stream>>>(
      Qh, Kh, Vh, mask, attnh);

  // Output projection (f32 out)
  gemm_f16_kernel<true, false, false><<<g768, 256, 0, stream>>>(
      attnh, Woh, bo, attno, nullptr, DM, DM);

  // x1 = LN(x + attn_out), plus f16 copy
  add_ln_kernel<<<MROWS, 256, 0, stream>>>(x, attno, g1, be1, x1, x1h);

  // FFN
  gemm_f16_kernel<false, true, true><<<gff, 256, 0, stream>>>(
      x1h, W1h, b1, nullptr, hh, DFF, DM);
  gemm_f16_kernel<true, false, false><<<g768, 256, 0, stream>>>(
      hh, W2h, b2, ff, nullptr, DM, DFF);

  // out = LN(x1 + ff)
  add_ln_kernel<<<MROWS, 256, 0, stream>>>(x1, ff, g2, be2, out, nullptr);
}
